// FusionLerpBlock_63909113364833
// MI455X (gfx1250) — matrix operand order repaired and checked by reading, not compiled, not run on hardware
//
#include <hip/hip_runtime.h>
#include <hip/hip_bf16.h>

#define BATCH 4
#define NPTS 100000
#define CDIM 128
#define INNER 48
#define HEADS 4
#define DHEAD 12
#define PEK 64                    // 51 padded to 64
#define TILE 16
#define TILES_PER_BATCH (NPTS / TILE)      // 6250
#define NTILES (BATCH * TILES_PER_BATCH)   // 25000
#define WPB 2                     // waves (tiles) per block

typedef float  v8f   __attribute__((ext_vector_type(8)));
typedef __bf16 v16bf __attribute__((ext_vector_type(16)));
typedef __bf16 v8bf  __attribute__((ext_vector_type(8)));
typedef __bf16 v4bf  __attribute__((ext_vector_type(4)));

// ---------------- fragment helpers (wave32, CDNA5 WMMA layouts) ----------------

__device__ __forceinline__ v16bf make16(v8bf lo, v8bf hi) {
    return __builtin_shufflevector(lo, hi, 0,1,2,3,4,5,6,7,8,9,10,11,12,13,14,15);
}

// A-matrix 16x32 bf16 from row-major LDS tile (row stride `stride` elements).
// ISA layout: lane L -> row M = L&15 ; lanes 0-15 hold K {0..7,16..23},
// lanes 16-31 hold K {8..15,24..31}  => two contiguous 16B LDS loads.
__device__ __forceinline__ v16bf load_a_lds(const __bf16* base, int stride,
                                            int kOff, int lane) {
    int m = lane & 15, ksel = lane >> 4;
    const __bf16* p = base + m * stride + kOff + ksel * 8;
    v8bf lo = *(const v8bf*)p;          // K = kOff+ksel*8 .. +7
    v8bf hi = *(const v8bf*)(p + 16);   // K = kOff+16+ksel*8 .. +7
    return make16(lo, hi);
}

// B-matrix 32x16 bf16 from a pre-TRANSPOSED global weight array wT[n][k]
// (row stride strideK elements). ISA layout: lane L -> col N = L&15,
// lanes 0-15 hold K {0..7,16..23} of the k-tile, lanes 16-31 hold K {8..15,24..31}
// => two 16B global loads per lane (L2 resident).
__device__ __forceinline__ v16bf load_b_gT(const __bf16* baseT, int strideK,
                                           int nOff, int kOff, int lane) {
    int n = lane & 15, half = lane >> 4;
    const __bf16* p = baseT + (size_t)(nOff + n) * strideK + kOff + half * 8;
    v8bf lo = *(const v8bf*)p;
    v8bf hi = *(const v8bf*)(p + 16);
    return make16(lo, hi);
}

__device__ __forceinline__ v8f wmma_bf16(v16bf a, v16bf b, v8f c) {
    return __builtin_amdgcn_wmma_f32_16x16x32_bf16(false, a, false, b,
                                                   (short)0, c, false, false);
}

// ---------------- kernel 1: time embedding MLP (4 rows, trivial) ----------------

__global__ void temb_kernel(const int* __restrict__ t,
                            const float* __restrict__ w1, const float* __restrict__ b1,
                            const float* __restrict__ w2, const float* __restrict__ b2,
                            float* __restrict__ temb) {
    __shared__ float ts[CDIM];
    __shared__ float hs[CDIM];
    int b = blockIdx.x, j = threadIdx.x;
    float tv = (float)t[b];
    float fr = __expf(-9.2103403719761836f * (float)(j & 63) * (1.0f / 64.0f));
    float ang = tv * fr;
    ts[j] = (j < 64) ? __sinf(ang) : __cosf(ang);
    __syncthreads();
    float acc = b1[j];
    for (int i = 0; i < CDIM; ++i) acc = fmaf(ts[i], w1[i * CDIM + j], acc);
    hs[j] = acc / (1.0f + __expf(-acc));            // silu
    __syncthreads();
    float acc2 = b2[j];
    for (int i = 0; i < CDIM; ++i) acc2 = fmaf(hs[i], w2[i * CDIM + j], acc2);
    temb[b * CDIM + j] = acc2;
}

// ------- kernel 2: convert weights to bf16, transposed [N][K], K zero-padded -------

#define PE_T_ELEMS   (CDIM * PEK)     // 8192
#define Q_T_ELEMS    (INNER * CDIM)   // 6144
#define KV_T_ELEMS   (96 * CDIM)      // 12288
#define OUT_T_ELEMS  (CDIM * PEK)     // 8192
#define PREP_TOTAL   (PE_T_ELEMS + Q_T_ELEMS + KV_T_ELEMS + OUT_T_ELEMS)

__global__ void prep_weights(const float* __restrict__ pe_w, const float* __restrict__ q_w,
                             const float* __restrict__ kv_w, const float* __restrict__ out_w,
                             __bf16* __restrict__ peT, __bf16* __restrict__ qT,
                             __bf16* __restrict__ kvT, __bf16* __restrict__ outT) {
    int i = blockIdx.x * blockDim.x + threadIdx.x;
    if (i < PE_T_ELEMS) {                       // pe_w (51,128) -> peT[128][64]
        int n = i >> 6, k = i & 63;
        peT[i] = (__bf16)((k < 51) ? pe_w[k * CDIM + n] : 0.0f);
        return;
    }
    i -= PE_T_ELEMS;
    if (i < Q_T_ELEMS) {                        // q_w (128,48) -> qT[48][128]
        int n = i >> 7, k = i & 127;
        qT[i] = (__bf16)q_w[k * INNER + n];
        return;
    }
    i -= Q_T_ELEMS;
    if (i < KV_T_ELEMS) {                       // kv_w (128,96) -> kvT[96][128]
        int n = i >> 7, k = i & 127;
        kvT[i] = (__bf16)kv_w[k * 96 + n];
        return;
    }
    i -= KV_T_ELEMS;
    if (i < OUT_T_ELEMS) {                      // out_w (48,128) -> outT[128][64]
        int n = i >> 6, k = i & 63;
        outT[i] = (__bf16)((k < 48) ? out_w[k * CDIM + n] : 0.0f);
    }
}

// ---------------- kernel 3: fused per-tile pipeline, one wave32 per 16 points ----------------

__global__ __launch_bounds__(WPB * 32)
void fused_kernel(const float* __restrict__ fT, const float* __restrict__ ft,
                  const float* __restrict__ pos,
                  const float* __restrict__ pe_b, const float* __restrict__ out_b,
                  const float* __restrict__ temb,
                  const __bf16* __restrict__ peT, const __bf16* __restrict__ qT,
                  const __bf16* __restrict__ kvT, const __bf16* __restrict__ outT,
                  float* __restrict__ out) {
    // per-wave LDS regions with lifetime overlays:
    //   regA (8KB):  aPE[16][64] -> aCTX[32][128] -> aO[16][64]
    //   regB (12KB): aEMB[16][128] -> kvS f32[32][96]
    //   regQ (3KB):  qS f32[16][48]
    constexpr int REGA = 8192, REGB = 12288, REGQ = 3072;
    constexpr int WAVE_BYTES = REGA + REGB + REGQ;          // 23552
    __shared__ __align__(16) unsigned char smem[WPB * WAVE_BYTES];

    const int lane = threadIdx.x & 31;
    const int wave = threadIdx.x >> 5;
    const int tile = blockIdx.x * WPB + wave;
    if (tile >= NTILES) return;                              // wave-uniform
    const int bIdx = tile / TILES_PER_BATCH;
    const int n0   = (tile % TILES_PER_BATCH) * TILE;

    unsigned char* wb = smem + wave * WAVE_BYTES;
    __bf16* regA = (__bf16*)wb;
    __bf16* aEMB = (__bf16*)(wb + REGA);
    float*  kvS  = (float*)(wb + REGA);
    float*  qS   = (float*)(wb + REGA + REGB);

    const float* fTrow  = fT  + ((size_t)bIdx * NPTS + n0) * CDIM;
    const float* ftrow  = ft  + ((size_t)bIdx * NPTS + n0) * CDIM;
    const float* posrow = pos + ((size_t)bIdx * NPTS + n0) * 3;

    // prefetch the fT/ft rows we will stream later (global_prefetch_b8)
    {
        const char* pf = (const char*)((lane < 16) ? fTrow : ftrow)
                       + (size_t)(lane & 15) * (CDIM * 4);
        __builtin_prefetch(pf, 0, 3);
        __builtin_prefetch(pf + 256, 0, 3);
    }

    // ---- stage 0: positional-encoding input  aPE[16][64] = [sin(24)|cos(24)|pos(3)|0pad]
    for (int idx = lane; idx < TILE * PEK; idx += 32) {
        int r = idx >> 6, c = idx & 63;
        float v = 0.0f;
        if (c < 48) {
            int cc   = (c < 24) ? c : c - 24;
            int comp = cc >> 3;
            float freq = 3.14159265358979323846f * (float)(1 << (cc & 7));
            float ang  = posrow[r * 3 + comp] * freq;
            v = (c < 24) ? __sinf(ang) : __cosf(ang);
        } else if (c < 51) {
            v = posrow[r * 3 + (c - 48)];
        }
        regA[idx] = (__bf16)v;
    }

    const int nLane = lane & 15, hLane = lane >> 4;

    // ---- stage 1: embed = aPE @ pe_w + pe_b + t_emb[b]   (16x64 @ 64x128)
    for (int nt = 0; nt < 8; ++nt) {
        v8f acc = {};
        #pragma unroll
        for (int kt = 0; kt < 2; ++kt) {
            v16bf a = load_a_lds(regA, PEK, kt * 32, lane);
            v16bf b = load_b_gT(peT, PEK, nt * 16, kt * 32, lane);
            acc = wmma_bf16(a, b, acc);
        }
        int col = nt * 16 + nLane;
        float bias = pe_b[col] + temb[bIdx * CDIM + col];
        #pragma unroll
        for (int j = 0; j < 8; ++j)
            aEMB[(j + hLane * 8) * CDIM + col] = (__bf16)(acc[j] + bias);
    }

    // ---- stage 2: q = embed @ q_w, pre-scaled by dh^-0.5   (16x128 @ 128x48)
    const float qscale = 0.28867513459481287f;   // 12^-0.5
    for (int nt = 0; nt < 3; ++nt) {
        v8f acc = {};
        #pragma unroll
        for (int kt = 0; kt < 4; ++kt) {
            v16bf a = load_a_lds(aEMB, CDIM, kt * 32, lane);
            v16bf b = load_b_gT(qT, CDIM, nt * 16, kt * 32, lane);
            acc = wmma_bf16(a, b, acc);
        }
        #pragma unroll
        for (int j = 0; j < 8; ++j)
            qS[(j + hLane * 8) * INNER + nt * 16 + nLane] = acc[j] * qscale;
    }

    // ---- stage 3: stage ctx rows (fT,ft interleaved) as bf16 A-tile aCTX[32][128]
    for (int idx = lane; idx < TILE * 32; idx += 32) {
        int p = idx >> 5, c4 = idx & 31;
        float4 a4 = ((const float4*)(fTrow + p * CDIM))[c4];
        float4 b4 = ((const float4*)(ftrow + p * CDIM))[c4];
        v4bf pa = {(__bf16)a4.x, (__bf16)a4.y, (__bf16)a4.z, (__bf16)a4.w};
        v4bf pb = {(__bf16)b4.x, (__bf16)b4.y, (__bf16)b4.z, (__bf16)b4.w};
        *(v4bf*)(regA + (2 * p)     * CDIM + c4 * 4) = pa;
        *(v4bf*)(regA + (2 * p + 1) * CDIM + c4 * 4) = pb;
    }

    // ---- stage 4: kv = ctx @ kv_w   (32x128 @ 128x96) -> kvS f32[32][96]
    for (int mt = 0; mt < 2; ++mt) {
        for (int nt = 0; nt < 6; ++nt) {
            v8f acc = {};
            #pragma unroll
            for (int kt = 0; kt < 4; ++kt) {
                v16bf a = load_a_lds(regA + mt * 16 * CDIM, CDIM, kt * 32, lane);
                v16bf b = load_b_gT(kvT, CDIM, nt * 16, kt * 32, lane);
                acc = wmma_bf16(a, b, acc);
            }
            #pragma unroll
            for (int j = 0; j < 8; ++j)
                kvS[(mt * 16 + j + hLane * 8) * 96 + nt * 16 + nLane] = acc[j];
        }
    }

    // ---- stage 5: per-(point,head) 2-key softmax attention -> aO[16][64] bf16 (padded)
    #pragma unroll
    for (int u = 0; u < 2; ++u) {
        int pair = u * 32 + lane;         // 64 (point,head) pairs
        int p = pair >> 2, h = pair & 3;
        const float* qp = qS  + p * INNER + h * DHEAD;
        const float* k0 = kvS + (2 * p)     * 96 + h * DHEAD;
        const float* k1 = kvS + (2 * p + 1) * 96 + h * DHEAD;
        float s0 = 0.0f, s1 = 0.0f;
        #pragma unroll
        for (int d = 0; d < DHEAD; ++d) {
            s0 = fmaf(qp[d], k0[d], s0);
            s1 = fmaf(qp[d], k1[d], s1);
        }
        float mx = fmaxf(s0, s1);
        float w0 = __expf(s0 - mx), w1 = __expf(s1 - mx);
        float inv = 1.0f / (w0 + w1);
        w0 *= inv; w1 *= inv;
        #pragma unroll
        for (int d = 0; d < DHEAD; ++d)
            regA[p * PEK + h * DHEAD + d] = (__bf16)fmaf(w0, k0[48 + d], w1 * k1[48 + d]);
    }
    for (int idx = lane; idx < TILE * 16; idx += 32)        // zero K-pad 48..63
        regA[(idx >> 4) * PEK + 48 + (idx & 15)] = (__bf16)0.0f;

    // ---- stage 6: gate = o @ out_w + out_b ; out = ft + gate*(fT-ft)
    float* outrow = out + ((size_t)bIdx * NPTS + n0) * CDIM;
    for (int nt = 0; nt < 8; ++nt) {
        v8f acc = {};
        #pragma unroll
        for (int kt = 0; kt < 2; ++kt) {
            v16bf a = load_a_lds(regA, PEK, kt * 32, lane);
            v16bf b = load_b_gT(outT, PEK, nt * 16, kt * 32, lane);
            acc = wmma_bf16(a, b, acc);
        }
        int col = nt * 16 + nLane;
        float ob = out_b[col];
        #pragma unroll
        for (int j = 0; j < 8; ++j) {
            int m = j + hLane * 8;
            float g  = acc[j] + ob;
            float vT = fTrow[m * CDIM + col];
            float vt = ftrow[m * CDIM + col];
            outrow[m * CDIM + col] = fmaf(g, vT - vt, vt);
        }
    }
}

// ---------------- launcher ----------------

extern "C" void kernel_launch(void* const* d_in, const int* in_sizes, int n_in,
                              void* d_out, int out_size, void* d_ws, size_t ws_size,
                              hipStream_t stream) {
    const float* fT    = (const float*)d_in[0];
    const float* ft    = (const float*)d_in[1];
    const float* pos   = (const float*)d_in[2];
    const int*   t     = (const int*)d_in[3];
    const float* pe_w  = (const float*)d_in[4];
    const float* pe_b  = (const float*)d_in[5];
    const float* t_w1  = (const float*)d_in[6];
    const float* t_b1  = (const float*)d_in[7];
    const float* t_w2  = (const float*)d_in[8];
    const float* t_b2  = (const float*)d_in[9];
    const float* q_w   = (const float*)d_in[10];
    const float* kv_w  = (const float*)d_in[11];
    const float* out_w = (const float*)d_in[12];
    const float* out_b = (const float*)d_in[13];
    float* out = (float*)d_out;

    // workspace layout (written every launch; deterministic)
    char* ws = (char*)d_ws;
    float*  temb = (float*)ws;                                   // 4*128 f32
    __bf16* peT  = (__bf16*)(ws + 2048);                         // [128][64]
    __bf16* qT   = (__bf16*)(ws + 2048 + PE_T_ELEMS * 2);        // [48][128]
    __bf16* kvT  = (__bf16*)(ws + 2048 + (PE_T_ELEMS + Q_T_ELEMS) * 2);
    __bf16* outT = (__bf16*)(ws + 2048 + (PE_T_ELEMS + Q_T_ELEMS + KV_T_ELEMS) * 2);

    temb_kernel<<<BATCH, CDIM, 0, stream>>>(t, t_w1, t_b1, t_w2, t_b2, temb);
    prep_weights<<<(PREP_TOTAL + 255) / 256, 256, 0, stream>>>(
        pe_w, q_w, kv_w, out_w, peT, qT, kvT, outT);
    fused_kernel<<<(NTILES + WPB - 1) / WPB, WPB * 32, 0, stream>>>(
        fT, ft, pos, pe_b, out_b, temb, peT, qT, kvT, outT, out);
}
